// RoFormerLayer_27169963114831
// MI455X (gfx1250) — matrix operand order repaired and checked by reading, not compiled, not run on hardware
//
#include <hip/hip_runtime.h>
#include <cstdint>
#include <cstddef>

// ---------------- problem constants ----------------
#define B_   4
#define S_   2048
#define HD_  1024
#define NH_  16
#define D_   64
#define FF_  4096
#define MR_  (B_ * S_)   // 8192 rows of activations

typedef unsigned short u16;
typedef __attribute__((ext_vector_type(16))) __bf16 v16bf;
typedef __attribute__((ext_vector_type(8)))  float  v8f;

union FragU {
    v16bf    h;
    uint32_t u[8];
    __device__ FragU() {}
};

__device__ __forceinline__ u16 f2bf(float f) {
    uint32_t x = __float_as_uint(f);
    uint32_t r = x + 0x7FFFu + ((x >> 16) & 1u);   // round-to-nearest-even
    return (u16)(r >> 16);
}

__device__ __forceinline__ v8f vzero8() {
    v8f z = {0.f, 0.f, 0.f, 0.f, 0.f, 0.f, 0.f, 0.f};
    return z;
}

__device__ __forceinline__ v8f wmma_bf16(const FragU& a, const FragU& b, v8f c) {
    return __builtin_amdgcn_wmma_f32_16x16x32_bf16(
        /*neg_a=*/false, a.h, /*neg_b=*/false, b.h,
        /*c_mod=*/(short)0, c, /*reuse_a=*/false, /*reuse_b=*/false);
}

__device__ __forceinline__ float gelu_tanh(float x) {
    float x3 = x * x * x;
    return 0.5f * x * (1.f + tanhf(0.7978845608028654f * (x + 0.044715f * x3)));
}

// LDS generic pointer -> raw 32-bit LDS offset (for async-to-LDS vdst operand)
__device__ __forceinline__ unsigned lds_addr32(const void* p) {
    typedef __attribute__((address_space(3))) const unsigned char lds_b;
    return (unsigned)(size_t)(lds_b*)p;
}

// CDNA5 async global->LDS 128-bit copy (ASYNCcnt-tracked); per-lane addresses.
__device__ __forceinline__ void async_ld_b128(unsigned lds_off, const void* gptr) {
    uint64_t ga = (uint64_t)gptr;
    asm volatile("global_load_async_to_lds_b128 %0, %1, off"
                 :: "v"(lds_off), "v"(ga) : "memory");
}
__device__ __forceinline__ void wait_asynccnt0() {
    asm volatile("s_wait_asynccnt 0x0" ::: "memory");
}

// ---------------- f32 -> bf16 convert ----------------
__global__ __launch_bounds__(256) void cvt_bf16_k(const float* __restrict__ in,
                                                  u16* __restrict__ out, size_t n) {
    size_t i = (size_t)blockIdx.x * blockDim.x + threadIdx.x;
    if (i < n) out[i] = f2bf(in[i]);
}

// ---------------- tiled WMMA GEMM: C[M,N] = A[M,K](bf16) @ W[K,N](bf16) + bias ----------------
enum { EPI_F32 = 0, EPI_RES_F32 = 1, EPI_GELU_BF16 = 2 };

#define BM 128
#define BN 128
#define BK 32

template <int EPI>
__global__ __launch_bounds__(256) void gemm_bf16_k(
    const u16* __restrict__ A, const u16* __restrict__ Bw,
    const float* __restrict__ bias, const float* __restrict__ res,
    void* __restrict__ out, int M, int N, int K)
{
    // double-buffered tiles: 2*(8KB + 8KB) = 32KB of the 320KB WGP pool
    __shared__ __align__(16) u16 lA[2][BM * BK];   // [row][k]
    __shared__ __align__(16) u16 lB[2][BN * BK];   // [n][k] (transposed)

    const int tid   = threadIdx.x;
    const int bm0   = blockIdx.y * BM;
    const int bn0   = blockIdx.x * BN;
    const int wave  = tid >> 5;
    const int lane  = tid & 31;
    const int wr    = wave & 3;     // 4 wave-rows of 32
    const int wc    = wave >> 2;    // 2 wave-cols of 64
    const int lh    = lane >> 4;    // half (0/1)
    const int l16   = lane & 15;

    const int nTiles = K / BK;

    v8f acc[2][4];
    #pragma unroll
    for (int mt = 0; mt < 2; ++mt)
        #pragma unroll
        for (int nt = 0; nt < 4; ++nt) acc[mt][nt] = vzero8();

    // per-thread staging coordinates (2 x 16B chunks each for A and B)
    int arow[2], ach[2], bkr[2], bnc[2];
    #pragma unroll
    for (int i = 0; i < 2; ++i) {
        int c = tid + i * 256;          // 0..511
        arow[i] = c >> 2;  ach[i] = c & 3;
        bkr[i]  = c >> 4;  bnc[i] = c & 15;
    }

    // ---- prologue: stage tile 0 into buffer 0
    #pragma unroll
    for (int i = 0; i < 2; ++i) {
        async_ld_b128(lds_addr32(&lA[0][arow[i] * BK + ach[i] * 8]),
                      A + (size_t)(bm0 + arow[i]) * K + ach[i] * 8);
    }
    {
        uint4 breg[2];
        #pragma unroll
        for (int i = 0; i < 2; ++i)
            breg[i] = *(const uint4*)(Bw + (size_t)bkr[i] * N + bn0 + bnc[i] * 8);
        #pragma unroll
        for (int i = 0; i < 2; ++i) {
            const u16* ds = (const u16*)&breg[i];
            #pragma unroll
            for (int j = 0; j < 8; ++j)
                lB[0][(bnc[i] * 8 + j) * BK + bkr[i]] = ds[j];
        }
    }
    wait_asynccnt0();
    __syncthreads();

    for (int t = 0; t < nTiles; ++t) {
        const int cur = t & 1, nxt = cur ^ 1;

        // ---- stage tile t+1 into the other buffer (overlaps WMMA below)
        if (t + 1 < nTiles) {
            const int k1 = (t + 1) * BK;
            #pragma unroll
            for (int i = 0; i < 2; ++i) {
                async_ld_b128(lds_addr32(&lA[nxt][arow[i] * BK + ach[i] * 8]),
                              A + (size_t)(bm0 + arow[i]) * K + k1 + ach[i] * 8);
            }
            uint4 breg[2];
            #pragma unroll
            for (int i = 0; i < 2; ++i)
                breg[i] = *(const uint4*)(Bw + (size_t)(k1 + bkr[i]) * N + bn0 + bnc[i] * 8);
            #pragma unroll
            for (int i = 0; i < 2; ++i) {
                const u16* ds = (const u16*)&breg[i];
                #pragma unroll
                for (int j = 0; j < 8; ++j)
                    lB[nxt][(bnc[i] * 8 + j) * BK + bkr[i]] = ds[j];
            }
        }
        if (t + 2 < nTiles)   // pull tile t+2's B lines into GL2 (192MB) ahead of time
            __builtin_prefetch(Bw + (size_t)((t + 2) * BK + bkr[0]) * N + bn0 + bnc[0] * 8, 0, 1);

        // ---- fragments (ISA 7.12.2 16-bit layouts) from current buffer
        FragU af[2], bf[4];
        #pragma unroll
        for (int mt = 0; mt < 2; ++mt) {
            int row = wr * 32 + mt * 16 + l16;
            #pragma unroll
            for (int v = 0; v < 8; ++v) {
                int kk = (v < 4 ? 2 * v : 16 + 2 * (v - 4)) + 8 * lh;
                af[mt].u[v] = *(const uint32_t*)(&lA[cur][row * BK + kk]);
            }
        }
        #pragma unroll
        for (int nt = 0; nt < 4; ++nt) {
            int n = wc * 64 + nt * 16 + l16;
            #pragma unroll
            for (int v = 0; v < 8; ++v) {
                int kk = (v < 4 ? 2 * v : 16 + 2 * (v - 4)) + 8 * lh;
                bf[nt].u[v] = *(const uint32_t*)(&lB[cur][n * BK + kk]);
            }
        }
        #pragma unroll
        for (int mt = 0; mt < 2; ++mt)
            #pragma unroll
            for (int nt = 0; nt < 4; ++nt)
                acc[mt][nt] = wmma_bf16(af[mt], bf[nt], acc[mt][nt]);

        wait_asynccnt0();     // tile t+1 async copies landed in LDS
        __syncthreads();      // + all waves done reading cur / writing nxt
    }

    // ---- epilogue
    #pragma unroll
    for (int mt = 0; mt < 2; ++mt)
        #pragma unroll
        for (int nt = 0; nt < 4; ++nt)
            #pragma unroll
            for (int r = 0; r < 8; ++r) {
                int row = bm0 + wr * 32 + mt * 16 + r + 8 * lh;
                int col = bn0 + wc * 64 + nt * 16 + l16;
                size_t idx = (size_t)row * N + col;
                float v = acc[mt][nt][r] + bias[col];
                if (EPI == EPI_F32) {
                    ((float*)out)[idx] = v;
                } else if (EPI == EPI_RES_F32) {
                    ((float*)out)[idx] = v + res[idx];
                } else {
                    ((u16*)out)[idx] = f2bf(gelu_tanh(v));
                }
            }
}

// ---------------- RoPE + head packing ----------------
// qf/kf/vf: fp32 [B,S,HD];  qh/kh: bf16 [B,NH,S,D];  vt: bf16 [B,NH,D,S]
__global__ __launch_bounds__(256) void rope_pack_k(
    const float* __restrict__ qf, const float* __restrict__ kf, const float* __restrict__ vf,
    u16* __restrict__ qh, u16* __restrict__ kh, u16* __restrict__ vt)
{
    size_t i = (size_t)blockIdx.x * blockDim.x + threadIdx.x;   // B*NH*S*(D/2) threads
    int d2 = (int)(i % (D_ / 2)); i /= (D_ / 2);
    int s  = (int)(i % S_);       i /= S_;
    int h  = (int)(i % NH_);
    int b  = (int)(i / NH_);
    int d0 = 2 * d2;

    float inv = powf(10000.f, -(float)d0 / (float)D_);
    float ang = (float)s * inv;
    float c = cosf(ang), sn = sinf(ang);

    size_t src = ((size_t)b * S_ + s) * HD_ + h * D_ + d0;
    float q0 = qf[src], q1 = qf[src + 1];
    float k0 = kf[src], k1 = kf[src + 1];
    float v0 = vf[src], v1 = vf[src + 1];

    size_t dsti = (((size_t)b * NH_ + h) * S_ + s) * D_ + d0;
    qh[dsti]     = f2bf(q0 * c - q1 * sn);
    qh[dsti + 1] = f2bf(q1 * c + q0 * sn);
    kh[dsti]     = f2bf(k0 * c - k1 * sn);
    kh[dsti + 1] = f2bf(k1 * c + k0 * sn);

    size_t vti = (((size_t)b * NH_ + h) * D_ + d0) * S_ + s;
    vt[vti]      = f2bf(v0);
    vt[vti + S_] = f2bf(v1);
}

// ---------------- flash attention: one wave per 16-query tile ----------------
__global__ __launch_bounds__(32) void flash_attn_k(
    const u16* __restrict__ qh, const u16* __restrict__ kh, const u16* __restrict__ vt,
    u16* __restrict__ ctx)
{
    __shared__ float sP[16 * 32];
    __shared__ float sAlpha[16];
    __shared__ float sL[16];

    const int lane = threadIdx.x;
    const int l16  = lane & 15;
    const int lh   = lane >> 4;
    const int q0   = blockIdx.x * 16;
    const int bh   = blockIdx.y;                 // b*NH + h
    const u16* Qp = qh + (size_t)bh * S_ * D_;
    const u16* Kp = kh + (size_t)bh * S_ * D_;
    const u16* Vp = vt + (size_t)bh * D_ * S_;

    // Q fragments for the two d-halves (A: 16x32 bf16)
    FragU qa[2];
    #pragma unroll
    for (int half = 0; half < 2; ++half)
        #pragma unroll
        for (int v = 0; v < 8; ++v) {
            int d = (v < 4 ? 2 * v : 16 + 2 * (v - 4)) + 8 * lh + 32 * half;
            qa[half].u[v] = *(const uint32_t*)(Qp + (size_t)(q0 + l16) * D_ + d);
        }

    v8f ctxAcc[4];
    #pragma unroll
    for (int dt = 0; dt < 4; ++dt) ctxAcc[dt] = vzero8();

    float m_i = -INFINITY, l_i = 0.f;            // valid in lanes 0..15 (row = lane)
    const float scale = 0.125f;                  // 1/sqrt(64)

    for (int kb = 0; kb < S_; kb += 32) {
        if (kb + 32 < S_) {                      // prefetch next KV block into GL2
            __builtin_prefetch(Kp + (size_t)(kb + 32 + l16) * D_, 0, 1);
            __builtin_prefetch(Vp + (size_t)l16 * S_ + kb + 32, 0, 1);
        }
        // ---- scores 16x32 = Q @ K^T (two 16-key subtiles, two d-halves each)
        v8f sc[2];
        #pragma unroll
        for (int st = 0; st < 2; ++st) {
            sc[st] = vzero8();
            #pragma unroll
            for (int dh = 0; dh < 2; ++dh) {
                FragU kbf;
                #pragma unroll
                for (int v = 0; v < 8; ++v) {
                    int d   = (v < 4 ? 2 * v : 16 + 2 * (v - 4)) + 8 * lh + 32 * dh;
                    int key = kb + st * 16 + l16;
                    kbf.u[v] = *(const uint32_t*)(Kp + (size_t)key * D_ + d);
                }
                sc[st] = wmma_bf16(qa[dh], kbf, sc[st]);
            }
        }
        #pragma unroll
        for (int st = 0; st < 2; ++st)
            #pragma unroll
            for (int r = 0; r < 8; ++r)
                sP[(r + 8 * lh) * 32 + st * 16 + l16] = sc[st][r];
        __syncthreads();

        // ---- online softmax, one row per lane 0..15
        if (lane < 16) {
            int row = lane;
            float mx = m_i;
            #pragma unroll 4
            for (int j = 0; j < 32; ++j) mx = fmaxf(mx, sP[row * 32 + j] * scale);
            float sum = 0.f;
            #pragma unroll 4
            for (int j = 0; j < 32; ++j) {
                float p = __expf(sP[row * 32 + j] * scale - mx);
                sP[row * 32 + j] = p;
                sum += p;
            }
            float alpha = __expf(m_i - mx);
            l_i = l_i * alpha + sum;
            m_i = mx;
            sAlpha[row] = alpha;
        }
        __syncthreads();

        // ---- rescale running context
        #pragma unroll
        for (int r = 0; r < 8; ++r) {
            float al = sAlpha[r + 8 * lh];
            #pragma unroll
            for (int dt = 0; dt < 4; ++dt) ctxAcc[dt][r] *= al;
        }

        // ---- P fragment (16x32 bf16 A layout) from LDS
        FragU pa;
        #pragma unroll
        for (int v = 0; v < 8; ++v) {
            int kk = (v < 4 ? 2 * v : 16 + 2 * (v - 4)) + 8 * lh;
            float p0 = sP[l16 * 32 + kk];
            float p1 = sP[l16 * 32 + kk + 1];
            pa.u[v] = (uint32_t)f2bf(p0) | ((uint32_t)f2bf(p1) << 16);
        }
        // ---- ctx += P @ V  (V transposed in memory: contiguous key pairs)
        #pragma unroll
        for (int dt = 0; dt < 4; ++dt) {
            FragU vb;
            #pragma unroll
            for (int v = 0; v < 8; ++v) {
                int key = kb + (v < 4 ? 2 * v : 16 + 2 * (v - 4)) + 8 * lh;
                int d   = dt * 16 + l16;
                vb.u[v] = *(const uint32_t*)(Vp + (size_t)d * S_ + key);
            }
            ctxAcc[dt] = wmma_bf16(pa, vb, ctxAcc[dt]);
        }
        __syncthreads();   // protect sP before next iteration's store
    }

    if (lane < 16) sL[lane] = l_i;
    __syncthreads();

    const int b = bh / NH_, h = bh % NH_;
    #pragma unroll
    for (int r = 0; r < 8; ++r) {
        float linv = 1.f / sL[r + 8 * lh];
        int srow = q0 + r + 8 * lh;
        #pragma unroll
        for (int dt = 0; dt < 4; ++dt) {
            int d = dt * 16 + l16;
            ctx[((size_t)b * S_ + srow) * HD_ + h * D_ + d] = f2bf(ctxAcc[dt][r] * linv);
        }
    }
}

// ---------------- LayerNorm over HD_=1024 (one block per row) ----------------
__global__ __launch_bounds__(256) void layernorm_k(
    const float* __restrict__ in, const float* __restrict__ g, const float* __restrict__ beta,
    float* __restrict__ outf, u16* __restrict__ outbf)
{
    __shared__ float red[256];
    const int tid = threadIdx.x;
    const float* xr = in + (size_t)blockIdx.x * HD_;

    float v[4];
    float s = 0.f;
    #pragma unroll
    for (int i = 0; i < 4; ++i) { v[i] = xr[tid + i * 256]; s += v[i]; }
    red[tid] = s;
    __syncthreads();
    for (int st = 128; st > 0; st >>= 1) {
        if (tid < st) red[tid] += red[tid + st];
        __syncthreads();
    }
    float mean = red[0] * (1.f / HD_);
    __syncthreads();

    float s2 = 0.f;
    #pragma unroll
    for (int i = 0; i < 4; ++i) { float d = v[i] - mean; s2 += d * d; }
    red[tid] = s2;
    __syncthreads();
    for (int st = 128; st > 0; st >>= 1) {
        if (tid < st) red[tid] += red[tid + st];
        __syncthreads();
    }
    float rstd = rsqrtf(red[0] * (1.f / HD_) + 1e-12f);

    float* of = outf + (size_t)blockIdx.x * HD_;
    #pragma unroll
    for (int i = 0; i < 4; ++i) {
        int c = tid + i * 256;
        float y = (v[i] - mean) * rstd * g[c] + beta[c];
        of[c] = y;
        if (outbf) outbf[(size_t)blockIdx.x * HD_ + c] = f2bf(y);
    }
}

// ---------------- host-side orchestration ----------------
extern "C" void kernel_launch(void* const* d_in, const int* in_sizes, int n_in,
                              void* d_out, int out_size, void* d_ws, size_t ws_size,
                              hipStream_t stream)
{
    (void)in_sizes; (void)n_in; (void)out_size; (void)ws_size;

    const float* x   = (const float*)d_in[0];
    const float* Wq  = (const float*)d_in[1];
    const float* bq  = (const float*)d_in[2];
    const float* Wk  = (const float*)d_in[3];
    const float* bk  = (const float*)d_in[4];
    const float* Wv  = (const float*)d_in[5];
    const float* bv  = (const float*)d_in[6];
    const float* Wo  = (const float*)d_in[7];
    const float* bo  = (const float*)d_in[8];
    const float* g1  = (const float*)d_in[9];
    const float* be1 = (const float*)d_in[10];
    const float* W1  = (const float*)d_in[11];
    const float* b1  = (const float*)d_in[12];
    const float* W2  = (const float*)d_in[13];
    const float* b2  = (const float*)d_in[14];
    const float* g2  = (const float*)d_in[15];
    const float* be2 = (const float*)d_in[16];
    float* out = (float*)d_out;

    char* w = (char*)d_ws;
    size_t off = 0;
    auto alloc = [&](size_t bytes) -> void* {
        void* p = w + off;
        off = (off + bytes + 255) & ~(size_t)255;
        return p;
    };

    const size_t NE  = (size_t)MR_ * HD_;       // 8M activations
    const size_t NHD = (size_t)B_ * NH_ * S_ * D_;

    u16*   xbf    = (u16*)  alloc(NE * 2);
    u16*   wqbf   = (u16*)  alloc((size_t)HD_ * HD_ * 2);
    u16*   wkbf   = (u16*)  alloc((size_t)HD_ * HD_ * 2);
    u16*   wvbf   = (u16*)  alloc((size_t)HD_ * HD_ * 2);
    u16*   wobf   = (u16*)  alloc((size_t)HD_ * HD_ * 2);
    u16*   w1bf   = (u16*)  alloc((size_t)HD_ * FF_ * 2);
    u16*   w2bf   = (u16*)  alloc((size_t)FF_ * HD_ * 2);
    float* qf     = (float*)alloc(NE * 4);
    float* kf     = (float*)alloc(NE * 4);
    float* vf     = (float*)alloc(NE * 4);
    u16*   qhp    = (u16*)  alloc(NHD * 2);
    u16*   khp    = (u16*)  alloc(NHD * 2);
    u16*   vtp    = (u16*)  alloc(NHD * 2);
    u16*   ctxbf  = (u16*)  alloc(NE * 2);
    float* t1     = (float*)alloc(NE * 4);      // attn_out + x
    float* hf     = (float*)alloc(NE * 4);      // LN1 output (f32)
    u16*   hbf    = (u16*)  alloc(NE * 2);      // LN1 output (bf16)
    u16*   ffn1   = (u16*)  alloc((size_t)MR_ * FF_ * 2);
    float* t2     = (float*)alloc(NE * 4);      // ffn + h

    auto cvt = [&](const float* src, u16* dst, size_t n) {
        cvt_bf16_k<<<dim3((unsigned)((n + 255) / 256)), dim3(256), 0, stream>>>(src, dst, n);
    };

    // 1. bf16 conversions
    cvt(x,  xbf,  NE);
    cvt(Wq, wqbf, (size_t)HD_ * HD_);
    cvt(Wk, wkbf, (size_t)HD_ * HD_);
    cvt(Wv, wvbf, (size_t)HD_ * HD_);
    cvt(Wo, wobf, (size_t)HD_ * HD_);
    cvt(W1, w1bf, (size_t)HD_ * FF_);
    cvt(W2, w2bf, (size_t)FF_ * HD_);

    // 2. QKV projections
    dim3 gQKV(HD_ / BN, MR_ / BM);
    gemm_bf16_k<EPI_F32><<<gQKV, 256, 0, stream>>>(xbf, wqbf, bq, nullptr, qf, MR_, HD_, HD_);
    gemm_bf16_k<EPI_F32><<<gQKV, 256, 0, stream>>>(xbf, wkbf, bk, nullptr, kf, MR_, HD_, HD_);
    gemm_bf16_k<EPI_F32><<<gQKV, 256, 0, stream>>>(xbf, wvbf, bv, nullptr, vf, MR_, HD_, HD_);

    // 3. RoPE + pack to heads (V transposed)
    {
        size_t nthr = (size_t)B_ * NH_ * S_ * (D_ / 2);
        rope_pack_k<<<dim3((unsigned)(nthr / 256)), 256, 0, stream>>>(qf, kf, vf, qhp, khp, vtp);
    }

    // 4. flash attention
    flash_attn_k<<<dim3(S_ / 16, B_ * NH_), 32, 0, stream>>>(qhp, khp, vtp, ctxbf);

    // 5. output projection + residual(x)
    gemm_bf16_k<EPI_RES_F32><<<gQKV, 256, 0, stream>>>(ctxbf, wobf, bo, x, t1, MR_, HD_, HD_);

    // 6. LayerNorm 1 -> h (f32 + bf16)
    layernorm_k<<<dim3(MR_), 256, 0, stream>>>(t1, g1, be1, hf, hbf);

    // 7. FFN up + GELU -> bf16
    dim3 gFF1(FF_ / BN, MR_ / BM);
    gemm_bf16_k<EPI_GELU_BF16><<<gFF1, 256, 0, stream>>>(hbf, w1bf, b1, nullptr, ffn1, MR_, FF_, HD_);

    // 8. FFN down + residual(h)
    gemm_bf16_k<EPI_RES_F32><<<gQKV, 256, 0, stream>>>(ffn1, w2bf, b2, hf, t2, MR_, HD_, FF_);

    // 9. LayerNorm 2 -> final output (f32)
    layernorm_k<<<dim3(MR_), 256, 0, stream>>>(t2, g2, be2, out, nullptr);
}
